// QuantumTransformerBlock_65481071407975
// MI455X (gfx1250) — compile-verified, matrix operand order checked by reading, not run on hardware
//
#include <hip/hip_runtime.h>
#include <hip/hip_bf16.h>

// Problem constants (must match reference)
#define B_ 2
#define S_ 2048
#define E_ 1024
#define H_ 16
#define D_ 64
#define F_ 1024
#define M_ (B_ * S_)          // 4096 rows of activations

// Attention LDS score-strip row stride (padded: 2048 + 4 to stagger banks)
#define SCP_ 2052

typedef __attribute__((ext_vector_type(16))) _Float16 v16h;
typedef __attribute__((ext_vector_type(8)))  _Float16 v8h;
typedef __attribute__((ext_vector_type(8)))  float    v8f;

union V16U { v16h v; v8h h[2]; };

// ---------------------------------------------------------------------------
// Fragment load: 16-bit A/B operand for v_wmma_f32_16x16x32_f16.
// tile points at row 0 of a 16-row tile; ld = row stride in halves.
// Per ISA: lanes 0-15 hold row (lane), halves = K{0..7} then K{16..23};
//          lanes 16-31 hold row (lane-16), halves = K{8..15} then K{24..31}.
// ---------------------------------------------------------------------------
__device__ __forceinline__ v16h load_frag(const _Float16* tile, int ld, int k0) {
    const int lane  = threadIdx.x & 31;
    const int row   = lane & 15;
    const int khalf = (lane >> 4) * 8;
    const _Float16* p = tile + (size_t)row * (size_t)ld + (size_t)(k0 + khalf);
    V16U u;
    u.h[0] = *(const v8h*)(p);
    u.h[1] = *(const v8h*)(p + 16);
    return u.v;
}

__device__ __forceinline__ v8f wmma_f16(v16h a, v16h b, v8f c) {
    return __builtin_amdgcn_wmma_f32_16x16x32_f16(false, a, false, b, (short)0, c, false, false);
}

// One wave computes a 64x64 output tile: 4x4 fragments, K in steps of 32.
// A row-major [.,K]; Bt row-major [.,K] (i.e. B pre-transposed).
__device__ __forceinline__ void gemm64x64(const _Float16* A, const _Float16* Bt,
                                          int K, v8f acc[4][4]) {
    const _Float16* arow[4];
    const _Float16* brow[4];
#pragma unroll
    for (int t = 0; t < 4; ++t) {
        arow[t] = A + (size_t)(t * 16) * K;
        brow[t] = Bt + (size_t)(t * 16) * K;
    }
#pragma unroll 1
    for (int k0 = 0; k0 < K; k0 += 32) {
        v16h af[4], bf[4];
#pragma unroll
        for (int t = 0; t < 4; ++t) af[t] = load_frag(arow[t], K, k0);
#pragma unroll
        for (int t = 0; t < 4; ++t) bf[t] = load_frag(brow[t], K, k0);
#pragma unroll
        for (int i = 0; i < 4; ++i)
#pragma unroll
            for (int j = 0; j < 4; ++j)
                acc[i][j] = wmma_f16(af[i], bf[j], acc[i][j]);
    }
}

__device__ __forceinline__ void zero_acc(v8f acc[4][4]) {
#pragma unroll
    for (int i = 0; i < 4; ++i)
#pragma unroll
        for (int j = 0; j < 4; ++j)
#pragma unroll
            for (int r = 0; r < 8; ++r) acc[i][j][r] = 0.0f;
}

// ---------------------------------------------------------------------------
// Elementwise helpers
// ---------------------------------------------------------------------------
__global__ void to_f16_kernel(const float* __restrict__ in, _Float16* __restrict__ out, int n) {
    int i = blockIdx.x * blockDim.x + threadIdx.x;
    if (i < n) out[i] = (_Float16)in[i];
}

// wT[n*K + k] = (f16) w[k*N + n]     (w is [K,N] row-major)
__global__ void transpose_f16_kernel(const float* __restrict__ w, _Float16* __restrict__ wT,
                                     int Kd, int Nd) {
    int i = blockIdx.x * blockDim.x + threadIdx.x;
    if (i >= Kd * Nd) return;
    int n = i / Kd;
    int k = i - n * Kd;
    wT[i] = (_Float16)w[(size_t)k * Nd + n];
}

// ---------------------------------------------------------------------------
// QKV GEMM: A = xh [M,E] f16, Bt = w{q,k,v}^T [E,E] f16.
// q,k stored [B,H,S,D] f16 (row-major over D); v stored transposed [B,H,D,S].
// ---------------------------------------------------------------------------
__global__ __launch_bounds__(128, 1)
void gemm_qkv_kernel(const _Float16* __restrict__ xh,
                     const _Float16* __restrict__ wqT,
                     const _Float16* __restrict__ wkT,
                     const _Float16* __restrict__ wvT,
                     _Float16* __restrict__ qh,
                     _Float16* __restrict__ kh,
                     _Float16* __restrict__ vT) {
    const int which = blockIdx.z;
    const _Float16* Bt = (which == 0) ? wqT : ((which == 1) ? wkT : wvT);
    const int m0 = (blockIdx.y * 4 + threadIdx.y) * 64;
    const int n0 = blockIdx.x * 64;
    v8f acc[4][4];
    zero_acc(acc);
    gemm64x64(xh + (size_t)m0 * E_, Bt + (size_t)n0 * E_, E_, acc);

    const int lane = threadIdx.x & 31;
    const int n_in = lane & 15;
    const int mBase = (lane >> 4) * 8;
#pragma unroll
    for (int i = 0; i < 4; ++i)
#pragma unroll
        for (int j = 0; j < 4; ++j)
#pragma unroll
            for (int r = 0; r < 8; ++r) {
                int m = m0 + i * 16 + mBase + r;
                int n = n0 + j * 16 + n_in;
                int b = m >> 11, s = m & (S_ - 1);
                int h = n >> 6,  d = n & (D_ - 1);
                _Float16 val = (_Float16)acc[i][j][r];
                if (which == 0)
                    qh[(((size_t)b * H_ + h) * S_ + s) * D_ + d] = val;
                else if (which == 1)
                    kh[(((size_t)b * H_ + h) * S_ + s) * D_ + d] = val;
                else
                    vT[(((size_t)b * H_ + h) * D_ + d) * S_ + s] = val;
            }
}

// Generic 64x64-per-wave GEMM with selectable epilogue.
// mode 0: out32[m*N + n] = acc
// mode 1: out32[m*N + n] = max(acc + bias[n], 0)
// mode 2: out32[m*N + n] = acc + bias[n]
__global__ __launch_bounds__(128, 1)
void gemm_f32out_kernel(const _Float16* __restrict__ A,
                        const _Float16* __restrict__ Bt,
                        const float* __restrict__ bias,
                        float* __restrict__ out32,
                        int K, int N, int mode) {
    const int m0 = (blockIdx.y * 4 + threadIdx.y) * 64;
    const int n0 = blockIdx.x * 64;
    v8f acc[4][4];
    zero_acc(acc);
    gemm64x64(A + (size_t)m0 * K, Bt + (size_t)n0 * K, K, acc);

    const int lane = threadIdx.x & 31;
    const int n_in = lane & 15;
    const int mBase = (lane >> 4) * 8;
#pragma unroll
    for (int i = 0; i < 4; ++i)
#pragma unroll
        for (int j = 0; j < 4; ++j)
#pragma unroll
            for (int r = 0; r < 8; ++r) {
                int m = m0 + i * 16 + mBase + r;
                int n = n0 + j * 16 + n_in;
                float v = acc[i][j][r];
                if (mode == 1) v = fmaxf(v + bias[n], 0.0f);
                else if (mode == 2) v = v + bias[n];
                out32[(size_t)m * N + n] = v;
            }
}

// ---------------------------------------------------------------------------
// quantum[b,h,s,d] = cumprod_d cos(q[b,h,s,d])   (one thread per (b,h,s))
// ---------------------------------------------------------------------------
__global__ void quantum_q_kernel(const _Float16* __restrict__ qh, float* __restrict__ qu) {
    int t = blockIdx.x * blockDim.x + threadIdx.x;
    if (t >= B_ * H_ * S_) return;
    const _Float16* p = qh + (size_t)t * D_;
    float* o = qu + (size_t)t * D_;
    float run = 1.0f;
#pragma unroll 8
    for (int i = 0; i < D_; ++i) {
        run *= __cosf((float)p[i]);
        o[i] = run;
    }
}

// ---------------------------------------------------------------------------
// Attention: one block = 4 waves (128 thr), handles 16 query rows of one (b,h).
// LDS: padded 16x2052 fp32 score strip (~131 KB) + 4x16x64 partial-out + stats
// (~148 KB total; 2 blocks/WGP within CDNA5's 320 KB).
//  Pass 1: waves split key tiles; scores = 0.125 * q@k^T via WMMA -> LDS
//  Pass 2: 8 threads/row compute row max & sum(exp) (in-wave shfl reduce)
//  Pass 3: waves split key range; partial P@V via WMMA -> LDS partials
//  Pass 4: combine partials, scale by 1/sum, mix quantum path, store f16.
// ---------------------------------------------------------------------------
__global__ __launch_bounds__(128, 1)
void attn_kernel(const _Float16* __restrict__ qh,
                 const _Float16* __restrict__ kh,
                 const _Float16* __restrict__ vT,
                 const float* __restrict__ quantum1,
                 const float* __restrict__ attn_qw,
                 _Float16* __restrict__ attnmix) {
    extern __shared__ char smem_raw[];
    float* sc     = (float*)smem_raw;          // [16][SCP_]
    float* pout   = sc + 16 * SCP_;            // [4][16][64]
    float* rowmax = pout + 4 * 16 * 64;        // [16]
    float* rowinv = rowmax + 16;               // [16]

    const int lane = threadIdx.x & 31;
    const int wave = threadIdx.y;              // 0..3
    const int b = blockIdx.z, h = blockIdx.y, qt = blockIdx.x;
    const int q0 = qt * 16;
    const int n_in = lane & 15;
    const int mBase = (lane >> 4) * 8;
    const int arow = lane & 15;                // A-fragment row for this lane

    const _Float16* qbase = qh + (((size_t)b * H_ + h) * S_ + q0) * D_;
    const _Float16* kbase = kh + (((size_t)b * H_ + h) * S_) * D_;
    const _Float16* vbase = vT + (((size_t)b * H_ + h) * D_) * S_;

    v16h aq0 = load_frag(qbase, D_, 0);
    v16h aq1 = load_frag(qbase, D_, 32);

    // ---- Pass 1: scores -> LDS (waves split the 128 key tiles) ----
#pragma unroll 1
    for (int kt = wave; kt < S_ / 16; kt += 4) {
        v8f c;
#pragma unroll
        for (int r = 0; r < 8; ++r) c[r] = 0.0f;
        v16h bk0 = load_frag(kbase + (size_t)(kt * 16) * D_, D_, 0);
        v16h bk1 = load_frag(kbase + (size_t)(kt * 16) * D_, D_, 32);
        c = wmma_f16(aq0, bk0, c);
        c = wmma_f16(aq1, bk1, c);
#pragma unroll
        for (int r = 0; r < 8; ++r)
            sc[(mBase + r) * SCP_ + kt * 16 + n_in] = c[r] * 0.125f;
    }
    __syncthreads();

    // ---- Pass 2: row max + sum(exp); 8 threads per row ----
    {
        const int t = wave * 32 + lane;        // 0..127
        const int row = t >> 3;                // 0..15
        const int sub = t & 7;                 // 0..7
        float mx = -3.0e38f;
        for (int k = 0; k < S_ / 8; ++k)
            mx = fmaxf(mx, sc[row * SCP_ + sub + 8 * k]);
#pragma unroll
        for (int off = 1; off < 8; off <<= 1)
            mx = fmaxf(mx, __shfl_xor(mx, off, 32));
        float sm = 0.0f;
        for (int k = 0; k < S_ / 8; ++k)
            sm += __expf(sc[row * SCP_ + sub + 8 * k] - mx);
#pragma unroll
        for (int off = 1; off < 8; off <<= 1)
            sm += __shfl_xor(sm, off, 32);
        if (sub == 0) { rowmax[row] = mx; rowinv[row] = 1.0f / sm; }
    }
    __syncthreads();

    // ---- Pass 3: partial P@V (waves split the 64 key blocks of 32) ----
    v8f acc[4];
#pragma unroll
    for (int j = 0; j < 4; ++j)
#pragma unroll
        for (int r = 0; r < 8; ++r) acc[j][r] = 0.0f;

    const float myMax = rowmax[arow];
    const int khalf = (lane >> 4) * 8;
#pragma unroll 1
    for (int kb = wave; kb < S_ / 32; kb += 4) {
        v16h pa;
#pragma unroll
        for (int i = 0; i < 8; ++i) {
            float s0 = sc[arow * SCP_ + kb * 32 + khalf + i];
            float s1 = sc[arow * SCP_ + kb * 32 + khalf + 16 + i];
            pa[i]     = (_Float16)__expf(s0 - myMax);
            pa[8 + i] = (_Float16)__expf(s1 - myMax);
        }
#pragma unroll
        for (int j = 0; j < 4; ++j) {
            v16h bv = load_frag(vbase + (size_t)(j * 16) * S_, S_, kb * 32);
            acc[j] = wmma_f16(pa, bv, acc[j]);
        }
    }
    // stash partials: pout[wave][row][d]
#pragma unroll
    for (int j = 0; j < 4; ++j)
#pragma unroll
        for (int r = 0; r < 8; ++r)
            pout[(wave * 16 + mBase + r) * 64 + j * 16 + n_in] = acc[j][r];
    __syncthreads();

    // ---- Pass 4: combine partials + epilogue (quantum mix) ----
    {
        const float aw = attn_qw[0];
        const float wgt = 1.0f / (1.0f + __expf(-aw));
        const int t = wave * 32 + lane;        // 0..127, 8 elems each
#pragma unroll
        for (int e0 = 0; e0 < 8; ++e0) {
            int e = t * 8 + e0;                // 0..1023
            int row = e >> 6;                  // 0..15
            int d = e & 63;
            float v = pout[(0 * 16 + row) * 64 + d]
                    + pout[(1 * 16 + row) * 64 + d]
                    + pout[(2 * 16 + row) * 64 + d]
                    + pout[(3 * 16 + row) * 64 + d];
            float cl = v * rowinv[row];
            int s_idx = q0 + row;
            float qv = quantum1[(((size_t)b * H_ + h) * S_ + s_idx) * D_ + d];
            float o = wgt * qv + (1.0f - wgt) * cl;
            attnmix[((size_t)b * S_ + s_idx) * E_ + h * D_ + d] = (_Float16)o;
        }
    }
}

// ---------------------------------------------------------------------------
// LayerNorm over E=1024 per row; one wave per row; lane handles 32 strided cols.
// out32 = LN(a + bres); optional f16 copy.
// ---------------------------------------------------------------------------
__global__ void layernorm_kernel(const float* __restrict__ a,
                                 const float* __restrict__ bres,
                                 const float* __restrict__ g,
                                 const float* __restrict__ beta,
                                 float* __restrict__ out32,
                                 _Float16* __restrict__ out16) {
    const int rowid = blockIdx.x;
    const int lane = threadIdx.x & 31;
    const float* ar = a + (size_t)rowid * E_;
    const float* br = bres + (size_t)rowid * E_;
    float vals[32];
    float s = 0.0f, ss = 0.0f;
#pragma unroll
    for (int i = 0; i < 32; ++i) {
        int c = lane + 32 * i;
        float v = ar[c] + br[c];
        vals[i] = v;
        s += v;
        ss += v * v;
    }
#pragma unroll
    for (int off = 1; off < 32; off <<= 1) {
        s  += __shfl_xor(s,  off, 32);
        ss += __shfl_xor(ss, off, 32);
    }
    const float mean = s * (1.0f / E_);
    const float var = ss * (1.0f / E_) - mean * mean;
    const float rs = rsqrtf(var + 1e-5f);
#pragma unroll
    for (int i = 0; i < 32; ++i) {
        int c = lane + 32 * i;
        float y = (vals[i] - mean) * rs * g[c] + beta[c];
        out32[(size_t)rowid * E_ + c] = y;
        if (out16) out16[(size_t)rowid * E_ + c] = (_Float16)y;
    }
}

// ---------------------------------------------------------------------------
// FFN quantum path: quantum2 = cumprod cos(x1) over E per row (wave prefix
// product), mixed with classical2 = relu(x1@w1+b1); writes f16 h buffer.
// ---------------------------------------------------------------------------
__global__ void quantum_ffn_mix_kernel(const float* __restrict__ x1,
                                       const float* __restrict__ classical2,
                                       const float* __restrict__ ffn_qw,
                                       _Float16* __restrict__ hbuf) {
    const int rowid = blockIdx.x;
    const int lane = threadIdx.x & 31;
    const float* xr = x1 + (size_t)rowid * E_;
    const int base = lane * 32;
    float loc[32];
    float run = 1.0f;
#pragma unroll
    for (int i = 0; i < 32; ++i) {
        run *= __cosf(xr[base + i]);
        loc[i] = run;
    }
    // inclusive prefix product of per-lane totals, then shift for exclusive
    float pref = run;
#pragma unroll
    for (int off = 1; off < 32; off <<= 1) {
        float o = __shfl_up(pref, off, 32);
        if (lane >= off) pref *= o;
    }
    float excl = __shfl_up(pref, 1, 32);
    if (lane == 0) excl = 1.0f;

    const float fw = ffn_qw[0];
    const float wgt = 1.0f / (1.0f + __expf(-fw));
    const float* cr = classical2 + (size_t)rowid * (size_t)F_;
    _Float16* hr = hbuf + (size_t)rowid * (size_t)F_;
#pragma unroll
    for (int i = 0; i < 32; ++i) {
        float qv = excl * loc[i];
        float hv = wgt * qv + (1.0f - wgt) * cr[base + i];
        hr[base + i] = (_Float16)hv;
    }
}

// ---------------------------------------------------------------------------
// Host-side orchestration
// ---------------------------------------------------------------------------
extern "C" void kernel_launch(void* const* d_in, const int* in_sizes, int n_in,
                              void* d_out, int out_size, void* d_ws, size_t ws_size,
                              hipStream_t stream) {
    const float* x       = (const float*)d_in[0];
    const float* wq      = (const float*)d_in[2];
    const float* wk      = (const float*)d_in[3];
    const float* wv      = (const float*)d_in[4];
    const float* wo      = (const float*)d_in[5];
    const float* attn_qw = (const float*)d_in[7];
    const float* w1      = (const float*)d_in[8];
    const float* b1      = (const float*)d_in[9];
    const float* w2      = (const float*)d_in[10];
    const float* b2      = (const float*)d_in[11];
    const float* ffn_qw  = (const float*)d_in[13];
    const float* ln1_g   = (const float*)d_in[14];
    const float* ln1_b   = (const float*)d_in[15];
    const float* ln2_g   = (const float*)d_in[16];
    const float* ln2_b   = (const float*)d_in[17];
    float* out = (float*)d_out;

    char* ws = (char*)d_ws;
    const size_t MB = 1024ull * 1024ull;
    _Float16* xh        = (_Float16*)(ws + 0 * MB);    // [M,E] f16       8 MB
    _Float16* wqT       = (_Float16*)(ws + 8 * MB);    // [E,E] f16       2 MB
    _Float16* wkT       = (_Float16*)(ws + 10 * MB);
    _Float16* wvT       = (_Float16*)(ws + 12 * MB);
    _Float16* woT       = (_Float16*)(ws + 14 * MB);
    _Float16* w1T       = (_Float16*)(ws + 16 * MB);
    _Float16* w2T       = (_Float16*)(ws + 18 * MB);
    _Float16* qh        = (_Float16*)(ws + 20 * MB);   // [B,H,S,D] f16   8 MB
    _Float16* kh        = (_Float16*)(ws + 28 * MB);
    _Float16* vT        = (_Float16*)(ws + 36 * MB);   // [B,H,D,S] f16
    float*    quantum1  = (float*)   (ws + 44 * MB);   // [B,H,S,D] f32  16 MB
    _Float16* attnmix   = (_Float16*)(ws + 60 * MB);   // [M,E] f16       8 MB
    float*    attnproj  = (float*)   (ws + 68 * MB);   // [M,E] f32      16 MB
    float*    x1        = (float*)   (ws + 84 * MB);   // [M,E] f32      16 MB
    _Float16* x1h       = (_Float16*)(ws + 100 * MB);  // [M,E] f16       8 MB
    float*    classical2= (float*)   (ws + 108 * MB);  // [M,F] f32      16 MB
    _Float16* hbuf      = (_Float16*)(ws + 124 * MB);  // [M,F] f16       8 MB
    float*    ffnout    = (float*)   (ws + 132 * MB);  // [M,E] f32      16 MB

    // 1. Convert activations + transpose-convert weights to f16
    to_f16_kernel<<<(M_ * E_) / 256, 256, 0, stream>>>(x, xh, M_ * E_);
    const int WN = E_ * E_;
    transpose_f16_kernel<<<WN / 256, 256, 0, stream>>>(wq, wqT, E_, E_);
    transpose_f16_kernel<<<WN / 256, 256, 0, stream>>>(wk, wkT, E_, E_);
    transpose_f16_kernel<<<WN / 256, 256, 0, stream>>>(wv, wvT, E_, E_);
    transpose_f16_kernel<<<WN / 256, 256, 0, stream>>>(wo, woT, E_, E_);
    transpose_f16_kernel<<<(E_ * F_) / 256, 256, 0, stream>>>(w1, w1T, E_, F_);
    transpose_f16_kernel<<<(F_ * E_) / 256, 256, 0, stream>>>(w2, w2T, F_, E_);

    dim3 gblk(32, 4);
    dim3 ggrd(E_ / 64, M_ / 256);  // (16, 16)

    // 2. QKV projections (z selects q/k/v)
    gemm_qkv_kernel<<<dim3(E_ / 64, M_ / 256, 3), gblk, 0, stream>>>(
        xh, wqT, wkT, wvT, qh, kh, vT);

    // 3. Quantum circuit on q
    quantum_q_kernel<<<(B_ * H_ * S_) / 256, 256, 0, stream>>>(qh, quantum1);

    // 4. Attention (4-wave blocks; padded score strip + partial buffer in LDS)
    const size_t attn_lds =
        (size_t)(16 * SCP_ + 4 * 16 * 64 + 32) * sizeof(float); // ~148 KB
    attn_kernel<<<dim3(S_ / 16, H_, B_), dim3(32, 4), attn_lds, stream>>>(
        qh, kh, vT, quantum1, attn_qw, attnmix);

    // 5. Output projection (wo)
    gemm_f32out_kernel<<<ggrd, gblk, 0, stream>>>(attnmix, woT, nullptr, attnproj, E_, E_, 0);

    // 6. LN1: x1 = LN(x + attnproj); also f16 copy for next GEMM
    layernorm_kernel<<<M_, 32, 0, stream>>>(x, attnproj, ln1_g, ln1_b, x1, x1h);

    // 7. FFN GEMM1: classical2 = relu(x1 @ w1 + b1)
    gemm_f32out_kernel<<<dim3(F_ / 64, M_ / 256), gblk, 0, stream>>>(
        x1h, w1T, b1, classical2, E_, F_, 1);

    // 8. quantum2 + mix -> hbuf (f16)
    quantum_ffn_mix_kernel<<<M_, 32, 0, stream>>>(x1, classical2, ffn_qw, hbuf);

    // 9. FFN GEMM2: ffnout = h @ w2 + b2
    gemm_f32out_kernel<<<dim3(E_ / 64, M_ / 256), gblk, 0, stream>>>(
        hbuf, w2T, b2, ffnout, F_, E_, 2);

    // 10. LN2 -> final output (fp32)
    layernorm_kernel<<<M_, 32, 0, stream>>>(x1, ffnout, ln2_g, ln2_b, out, nullptr);

    (void)in_sizes; (void)n_in; (void)out_size; (void)ws_size;
}
